// S6_67310727463501
// MI455X (gfx1250) — matrix operand order repaired and checked by reading, not compiled, not run on hardware
//
#include <hip/hip_runtime.h>
#include <math.h>

#define D_MODEL 768
#define N_STATE 16
#define M_ROWS  8192   // B_SZ * SEQ = 4 * 2048

typedef __attribute__((ext_vector_type(16))) __bf16 v16bf;
typedef __attribute__((ext_vector_type(8)))  float  v8f;
typedef int v4i_ls __attribute__((vector_size(16)));   // matches async-LDS builtin param

struct Frag32 { uint4 lo; uint4 hi; };   // 32 bytes == 16 bf16 values

// round-to-nearest-even f32 -> bf16 (bit pattern as ushort)
__device__ __forceinline__ unsigned short f32_to_bf16(float f) {
    unsigned u = __float_as_uint(f);
    u += 0x7FFFu + ((u >> 16) & 1u);
    return (unsigned short)(u >> 16);
}

// numerically stable softplus, matches jax.nn.softplus
__device__ __forceinline__ float softplus(float z) {
    return fmaxf(z, 0.0f) + log1pf(expf(-fabsf(z)));
}

// ---------------------------------------------------------------------------
// Async global -> LDS copy of 16 bytes (per lane), tracked by ASYNCcnt.
// ---------------------------------------------------------------------------
__device__ __forceinline__ void async_copy_b128(const unsigned short* g,
                                                unsigned short* l) {
#if __has_builtin(__builtin_amdgcn_global_load_async_to_lds_b128)
    __builtin_amdgcn_global_load_async_to_lds_b128(
        (__attribute__((address_space(1))) v4i_ls*)g,
        (__attribute__((address_space(3))) v4i_ls*)l, 0, 0);
#else
    asm volatile("global_load_async_to_lds_b128 %0, %1, off"
                 :: "v"((__attribute__((address_space(3))) unsigned short*)l),
                    "v"((const __attribute__((address_space(1))) unsigned short*)g)
                 : "memory");
#endif
}

__device__ __forceinline__ void wait_async_le3() {
#if __has_builtin(__builtin_amdgcn_s_wait_asynccnt)
    __builtin_amdgcn_s_wait_asynccnt((short)3);
#else
    asm volatile("s_wait_asynccnt 0x3" ::: "memory");
#endif
}
__device__ __forceinline__ void wait_async_le0() {
#if __has_builtin(__builtin_amdgcn_s_wait_asynccnt)
    __builtin_amdgcn_s_wait_asynccnt((short)0);
#else
    asm volatile("s_wait_asynccnt 0x0" ::: "memory");
#endif
}

// ---------------------------------------------------------------------------
// Kernel 1: W1 [K=768, N=768] f32 row-major  ->  W1T [N=768, K=768] bf16
// ---------------------------------------------------------------------------
__global__ __launch_bounds__(256) void k_transpose_w1(
    const float* __restrict__ W1, unsigned short* __restrict__ W1T)
{
    int idx = blockIdx.x * 256 + threadIdx.x;        // over 768*768 outputs
    int n = idx / D_MODEL;
    int k = idx - n * D_MODEL;
    W1T[idx] = f32_to_bf16(W1[k * D_MODEL + n]);
}

// ---------------------------------------------------------------------------
// Kernel 2: per row r:  s[r] = dot(x_r@W2 + b2, x_r@W3 + b3)   (n = 16)
//           and convert x row to bf16 (each element exactly once).
// One wave (32 lanes) per row; lane owns 24 consecutive k values.
// ---------------------------------------------------------------------------
__global__ __launch_bounds__(256) void k_proj(
    const float* __restrict__ x,
    const float* __restrict__ W2, const float* __restrict__ b2,
    const float* __restrict__ W3, const float* __restrict__ b3,
    unsigned short* __restrict__ xbf, float* __restrict__ s)
{
    const int row  = blockIdx.x * 8 + (threadIdx.x >> 5);
    const int lane = threadIdx.x & 31;
    const float* xr = x + (size_t)row * D_MODEL;

    float accB[N_STATE], accC[N_STATE];
#pragma unroll
    for (int n = 0; n < N_STATE; ++n) { accB[n] = 0.0f; accC[n] = 0.0f; }

    const int kbase = lane * (D_MODEL / 32);         // 24 k's per lane
#pragma unroll 4
    for (int i = 0; i < D_MODEL / 32; ++i) {
        const int k = kbase + i;
        const float xv = xr[k];
        xbf[(size_t)row * D_MODEL + k] = f32_to_bf16(xv);
        const float4* w2v = (const float4*)(W2 + k * N_STATE);
        const float4* w3v = (const float4*)(W3 + k * N_STATE);
#pragma unroll
        for (int q = 0; q < 4; ++q) {
            const float4 a = w2v[q];
            const float4 c = w3v[q];
            accB[4*q+0] = fmaf(xv, a.x, accB[4*q+0]);
            accB[4*q+1] = fmaf(xv, a.y, accB[4*q+1]);
            accB[4*q+2] = fmaf(xv, a.z, accB[4*q+2]);
            accB[4*q+3] = fmaf(xv, a.w, accB[4*q+3]);
            accC[4*q+0] = fmaf(xv, c.x, accC[4*q+0]);
            accC[4*q+1] = fmaf(xv, c.y, accC[4*q+1]);
            accC[4*q+2] = fmaf(xv, c.z, accC[4*q+2]);
            accC[4*q+3] = fmaf(xv, c.w, accC[4*q+3]);
        }
    }

    // wave32 butterfly reduction across lanes
#pragma unroll
    for (int off = 16; off >= 1; off >>= 1) {
#pragma unroll
        for (int n = 0; n < N_STATE; ++n) {
            accB[n] += __shfl_xor(accB[n], off, 32);
            accC[n] += __shfl_xor(accC[n], off, 32);
        }
    }

    if (lane == 0) {
        float sv = 0.0f;
#pragma unroll
        for (int n = 0; n < N_STATE; ++n)
            sv = fmaf(accB[n] + b2[n], accC[n] + b3[n], sv);
        s[row] = sv;
    }
}

// ---------------------------------------------------------------------------
// Kernel 3: fused GEMM + epilogue, async-LDS double-buffered pipeline.
//   logits = xbf @ W1T^T   (M=8192, N=768, K=768), bf16 in / f32 accumulate
//   y[r,c] = x[r,c] * softplus(logits[r,c] + b1[c]) * s[r]
//
// Block tile: 64 rows x 128 cols, 8 waves (4 M-waves x 2 N-waves),
// each wave a 16x64 subtile via 4 accumulators sharing one A fragment.
// K pipelined in steps of 32 through double-buffered LDS filled with
// global_load_async_to_lds_b128 (ASYNCcnt), 3 async ops per wave per stage.
// LDS tiles padded to 40-half (80 B = 20 dword) stride to spread banks.
// ---------------------------------------------------------------------------
#define KSTEP   32
#define NSTEPS  (D_MODEL / KSTEP)   // 24
#define LDS_PAD 40                  // halves per row/col slot

__global__ __launch_bounds__(256) void k_gemm_fused(
    const unsigned short* __restrict__ xbf,
    const unsigned short* __restrict__ w1t,
    const float* __restrict__ x,
    const float* __restrict__ b1,
    const float* __restrict__ s,
    float* __restrict__ y)
{
    __shared__ unsigned short ldsA[2][64  * LDS_PAD];   // 10240 B
    __shared__ unsigned short ldsB[2][128 * LDS_PAD];   // 20480 B

    const int tid  = threadIdx.x;
    const int lane = tid & 31;
    const int wave = tid >> 5;
    const int mw   = wave >> 1;          // 0..3
    const int nw   = wave & 1;           // 0..1
    const int tN   = blockIdx.x % (D_MODEL / 128);   // 6 col-blocks of 128
    const int tM   = blockIdx.x / (D_MODEL / 128);   // 128 row-blocks of 64

    // ---- async tile copy: A 64x32 (4 KB, 256 x b128), B 128x32 (8 KB, 512 x b128)
    const int ar = tid >> 2;                 // A row 0..63
    const int ah = (tid & 3) * 8;            // A k-halves 0/8/16/24
    const unsigned short* gA0 = xbf + (size_t)(tM * 64 + ar) * D_MODEL + ah;
    unsigned short* lA0 = &ldsA[0][ar * LDS_PAD + ah];
    unsigned short* lA1 = &ldsA[1][ar * LDS_PAD + ah];

    const int bc0 = tid >> 1;                // B col 0..127 (part 0)
    const int bh0 = (tid & 1) * 8;           // k-halves 0/8
    const int bc1 = bc0;                     // part 1: same col, k-halves +16
    const int bh1 = bh0 + 16;
    const unsigned short* gB0 = w1t + (size_t)(tN * 128 + bc0) * D_MODEL + bh0;
    const unsigned short* gB1 = w1t + (size_t)(tN * 128 + bc1) * D_MODEL + bh1;
    unsigned short* lB0a = &ldsB[0][bc0 * LDS_PAD + bh0];
    unsigned short* lB0b = &ldsB[0][bc1 * LDS_PAD + bh1];
    unsigned short* lB1a = &ldsB[1][bc0 * LDS_PAD + bh0];
    unsigned short* lB1b = &ldsB[1][bc1 * LDS_PAD + bh1];

    // ---- fragment addressing inside the LDS tile
    // A (16-bit A 16x32): lanes 0-15: M=lane, K 0..15; lanes 16-31: K 16..31
    const int fr  = mw * 16 + (lane & 15);   // row in block tile
    const int fk  = (lane >> 4) * 16;        // k base of low half
    // B (32x16): lanes 0-15: N=lane, K 0..15; lanes 16-31: N=lane-16, K 16..31
    const int fc  = nw * 64 + (lane & 15);   // col of j=0 subtile
    const int fkb = (lane >> 4) * 16;

    // ---- prologue: prefetch stages 0 and 1
    async_copy_b128(gA0, lA0);
    async_copy_b128(gB0, lB0a);
    async_copy_b128(gB1, lB0b);
    async_copy_b128(gA0 + KSTEP, lA1);
    async_copy_b128(gB0 + KSTEP, lB1a);
    async_copy_b128(gB1 + KSTEP, lB1b);

    v8f acc0 = {}, acc1 = {}, acc2 = {}, acc3 = {};

    for (int step = 0; step < NSTEPS; ++step) {
        const int buf = step & 1;
        // newest 3 async ops (next stage) may stay in flight; current stage done
        if (step + 1 < NSTEPS) wait_async_le3();
        else                   wait_async_le0();
        __syncthreads();

        // A fragment from LDS
        Frag32 fa;
        fa.lo = *(const uint4*)&ldsA[buf][fr * LDS_PAD + fk];
        fa.hi = *(const uint4*)&ldsA[buf][fr * LDS_PAD + fk + 8];
        const v16bf a = __builtin_bit_cast(v16bf, fa);

        Frag32 fb;
        fb.lo = *(const uint4*)&ldsB[buf][(fc)      * LDS_PAD + fkb];
        fb.hi = *(const uint4*)&ldsB[buf][(fc)      * LDS_PAD + fkb + 8];
        const v16bf b0v = __builtin_bit_cast(v16bf, fb);
        fb.lo = *(const uint4*)&ldsB[buf][(fc + 16) * LDS_PAD + fkb];
        fb.hi = *(const uint4*)&ldsB[buf][(fc + 16) * LDS_PAD + fkb + 8];
        const v16bf b1v = __builtin_bit_cast(v16bf, fb);
        fb.lo = *(const uint4*)&ldsB[buf][(fc + 32) * LDS_PAD + fkb];
        fb.hi = *(const uint4*)&ldsB[buf][(fc + 32) * LDS_PAD + fkb + 8];
        const v16bf b2v = __builtin_bit_cast(v16bf, fb);
        fb.lo = *(const uint4*)&ldsB[buf][(fc + 48) * LDS_PAD + fkb];
        fb.hi = *(const uint4*)&ldsB[buf][(fc + 48) * LDS_PAD + fkb + 8];
        const v16bf b3v = __builtin_bit_cast(v16bf, fb);

        acc0 = __builtin_amdgcn_wmma_f32_16x16x32_bf16(false, a, false, b0v,
                                                       (short)0, acc0, false, false);
        acc1 = __builtin_amdgcn_wmma_f32_16x16x32_bf16(false, a, false, b1v,
                                                       (short)0, acc1, false, false);
        acc2 = __builtin_amdgcn_wmma_f32_16x16x32_bf16(false, a, false, b2v,
                                                       (short)0, acc2, false, false);
        acc3 = __builtin_amdgcn_wmma_f32_16x16x32_bf16(false, a, false, b3v,
                                                       (short)0, acc3, false, false);

        __syncthreads();   // everyone done reading buf before it is refilled

        if (step + 2 < NSTEPS) {
            const int kk = (step + 2) * KSTEP;
            if (buf == 0) {
                async_copy_b128(gA0 + kk, lA0);
                async_copy_b128(gB0 + kk, lB0a);
                async_copy_b128(gB1 + kk, lB0b);
            } else {
                async_copy_b128(gA0 + kk, lA1);
                async_copy_b128(gB0 + kk, lB1a);
                async_copy_b128(gB1 + kk, lB1b);
            }
        }
    }

    // ---- epilogue: y = x * softplus(acc + b1) * s[row]
    // C/D layout: VGPR i <-> row (lane<16 ? i : 8+i), col = lane&15
    const int rbase = tM * 64 + mw * 16 + ((lane >> 4) << 3);
    float sv[8];
#pragma unroll
    for (int i = 0; i < 8; ++i) sv[i] = s[rbase + i];

    v8f accs[4] = {acc0, acc1, acc2, acc3};
#pragma unroll
    for (int j = 0; j < 4; ++j) {
        const int c  = tN * 128 + nw * 64 + j * 16 + (lane & 15);
        const float bc = b1[c];
#pragma unroll
        for (int i = 0; i < 8; ++i) {
            const int r = rbase + i;
            const float z  = accs[j][i] + bc;
            const float xv = x[(size_t)r * D_MODEL + c];
            y[(size_t)r * D_MODEL + c] = xv * softplus(z) * sv[i];
        }
    }
}

// ---------------------------------------------------------------------------
extern "C" void kernel_launch(void* const* d_in, const int* in_sizes, int n_in,
                              void* d_out, int out_size, void* d_ws, size_t ws_size,
                              hipStream_t stream) {
    const float* x  = (const float*)d_in[0];
    const float* W1 = (const float*)d_in[1];
    const float* b1 = (const float*)d_in[2];
    const float* W2 = (const float*)d_in[3];
    const float* b2 = (const float*)d_in[4];
    const float* W3 = (const float*)d_in[5];
    const float* b3 = (const float*)d_in[6];
    // d_in[7] (A) is mathematically dead: h0 == 0 kills the dA*h term.
    float* y = (float*)d_out;

    // Workspace layout (~13.8 MB):
    //   xbf : M_ROWS*D_MODEL bf16   (12.58 MB)
    //   w1t : D_MODEL*D_MODEL bf16  ( 1.18 MB)
    //   s   : M_ROWS f32            ( 32 KB )
    unsigned short* xbf = (unsigned short*)d_ws;
    unsigned short* w1t = xbf + (size_t)M_ROWS * D_MODEL;
    float*          sbuf = (float*)(w1t + (size_t)D_MODEL * D_MODEL);

    k_transpose_w1<<<(D_MODEL * D_MODEL) / 256, 256, 0, stream>>>(W1, w1t);
    k_proj<<<M_ROWS / 8, 256, 0, stream>>>(x, W2, b2, W3, b3, xbf, sbuf);
    k_gemm_fused<<<(M_ROWS / 64) * (D_MODEL / 128), 256, 0, stream>>>(
        xbf, w1t, x, b1, sbuf, y);
}
